// EnsembleModel_34437047779754
// MI455X (gfx1250) — matrix operand order repaired and checked by reading, not compiled, not run on hardware
//
#include <hip/hip_runtime.h>

// ---------------------------------------------------------------------------
// Ensemble MLP (M=8, B=4096, IN=96, H=1024) on gfx1250 via bf16 WMMA.
// Phase 1: pack fp32 weights -> bf16 in WMMA-B tile order [m][K/32][Npad][32].
// Phase 2: GEMM kernels: triple-buffered LDS, depth-2 async global->LDS
//          pipeline (GLOBAL_LOAD_ASYNC_TO_LDS_B128 + counted s_wait_asynccnt),
//          v_wmma_f32_16x16x32_bf16 core, fused bias+tanh / log-var clamp.
// ---------------------------------------------------------------------------

#define M_ENS   8
#define BATCH   4096
#define OBS_D   64
#define ACT_D   32
#define IN_D    96
#define HID     1024

typedef __attribute__((ext_vector_type(16))) __bf16 v16bf;
typedef __attribute__((ext_vector_type(8)))  float  v8f;
typedef __attribute__((ext_vector_type(4)))  int    vint4;   // clang vector (not HIP struct)

typedef __attribute__((address_space(1))) vint4 gvint4;      // global
typedef __attribute__((address_space(3))) vint4 lvint4;      // LDS

enum { ACT_TANH = 0, ACT_ID = 1, ACT_CLAMP = 2 };

#if __has_builtin(__builtin_amdgcn_global_load_async_to_lds_b128) && \
    __has_builtin(__builtin_amdgcn_s_wait_asynccnt)
#define USE_ASYNC 1
#else
#define USE_ASYNC 0
#endif

__device__ __forceinline__ unsigned short f2bf(float f) {
  unsigned int x = __float_as_uint(f);
  x += 0x7FFFu + ((x >> 16) & 1u);          // round-to-nearest-even
  return (unsigned short)(x >> 16);
}

__device__ __forceinline__ float softplus_f(float x) {
  return (x > 20.0f) ? x : log1pf(__expf(x));
}

// one 16-byte global -> LDS copy (async on CDNA5, sync fallback otherwise)
__device__ __forceinline__ void cp16(unsigned short* dst_lds,
                                     const unsigned short* src) {
#if USE_ASYNC
  __builtin_amdgcn_global_load_async_to_lds_b128(
      (gvint4*)(vint4*)const_cast<unsigned short*>(src),
      (lvint4*)(vint4*)dst_lds, 0, 0);
#else
  *(int4*)dst_lds = *(const int4*)src;
#endif
}

template <int N>
__device__ __forceinline__ void drain_async_n() {
#if USE_ASYNC
  __builtin_amdgcn_s_wait_asynccnt(N);
#endif
}

union Frag { v16bf v; int4 q[2]; };
union Acc  { v8f  v; float f[8]; };

// ---------------------------------------------------------------------------
__global__ void prep_x_kernel(const float* __restrict__ obs,
                              const float* __restrict__ act,
                              unsigned short* __restrict__ xbf) {
  int i = blockIdx.x * blockDim.x + threadIdx.x;      // over B*IN_D
  if (i < BATCH * IN_D) {
    int b = i / IN_D, j = i - b * IN_D;
    float v = (j < OBS_D) ? obs[b * OBS_D + j] : act[b * ACT_D + (j - OBS_D)];
    xbf[i] = f2bf(v);
  }
}

// fp32 W[m][K][N] -> bf16 Wp[m][K/32][Npad][32]; zero-fill n >= N.
// One thread packs the 32 K-values of one (m, kc, n) row: strided fp32 reads
// (coalesced across n), one contiguous 64B write.
__global__ void pack_w_kernel(const float* __restrict__ W,
                              unsigned short* __restrict__ Wp,
                              int K, int N, int Npad) {
  int idx = blockIdx.x * blockDim.x + threadIdx.x;
  int KC = K >> 5;
  int total = M_ENS * KC * Npad;
  if (idx >= total) return;
  int n  = idx % Npad;
  int mk = idx / Npad;
  int kc = mk % KC;
  int m  = mk / KC;

  union { unsigned short s[32]; int4 q[4]; } u;
  if (n < N) {
    const float* src = W + ((size_t)m * K + (size_t)kc * 32) * N + n;
#pragma unroll
    for (int kk = 0; kk < 32; kk++) u.s[kk] = f2bf(src[(size_t)kk * N]);
  } else {
#pragma unroll
    for (int kk = 0; kk < 32; kk++) u.s[kk] = 0;
  }
  int4* dst = (int4*)(Wp + (size_t)idx * 32);
#pragma unroll
  for (int j = 0; j < 4; j++) dst[j] = u.q[j];
}

// ---------------------------------------------------------------------------
// C[m] = act( A[m](BATCH x K) * W[m](K x N) + bias[m] )
// A: bf16 row-major [BATCH][K].  Wp: bf16 packed [K/32][Wn][32] per ensemble.
// 256 threads = 8 waves: 4 row-groups (32 rows) x 2 col-groups (NT*16 cols).
// Depth-2 pipeline: iteration c stages chunk c+2; counted async wait retires
// chunk c+1's copies while leaving chunk c+2's in flight across the WMMAs.
template <int NT, int ACTK, bool OUT_BF16>
__global__ __launch_bounds__(256)
void gemm_wmma_kernel(const unsigned short* __restrict__ A, size_t a_mstride,
                      const unsigned short* __restrict__ Wp,
                      const float* __restrict__ bias,
                      void* __restrict__ outp,
                      int K, int N, int Wn) {
  constexpr int BM  = 128;
  constexpr int BN  = 32 * NT;
  constexpr int LDA = 40;   // shorts per LDS row: 32 data + 8 pad (b128-aligned, bank-friendly)
  constexpr int LDW = 40;
  constexpr int WSEG = BN * 4;           // 16B segments in a W tile
  // min per-wave async instructions issued per chunk (A: 2, W: NT-dependent)
  constexpr int CNT = 2 + (NT == 4 ? 2 : (NT == 2 ? 1 : 0));

  __shared__ __align__(16) unsigned short A_lds[3][BM * LDA];
  __shared__ __align__(16) unsigned short W_lds[3][BN * LDW];

  const int t    = threadIdx.x;
  const int lane = t & 31;
  const int wave = t >> 5;
  const int hl   = lane >> 4;
  const int l15  = lane & 15;

  const int m      = blockIdx.z;
  const int rowBlk = blockIdx.x * BM;
  const int colBlk = blockIdx.y * BN;

  const unsigned short* Am = A  + (size_t)m * a_mstride;
  const unsigned short* Wm = Wp + (size_t)m * (size_t)K * (size_t)Wn;
  const float* bm = bias + (size_t)m * (size_t)N;

  const int r_off = (wave >> 1) * 32;
  const int c_off = (wave & 1) * (NT * 16);

  // staging: A tile BM x 32 bf16 (512 x 16B segs), W tile contiguous BN*64B
  auto stageA = [&](int buf, int k0) {
#pragma unroll
    for (int i = 0; i < 2; i++) {
      int idx = t + i * 256;
      int row = idx >> 2;
      int seg = idx & 3;
      cp16(&A_lds[buf][row * LDA + seg * 8],
           &Am[(size_t)(rowBlk + row) * (size_t)K + k0 + seg * 8]);
    }
  };
  auto stageW = [&](int buf, int kc) {
    const unsigned short* src = Wm + ((size_t)kc * Wn + colBlk) * 32;
#pragma unroll
    for (int i = 0; i < (WSEG + 255) / 256; i++) {
      int idx = t + i * 256;
      if (WSEG >= 256 * (i + 1) || idx < WSEG)
        cp16(&W_lds[buf][(idx >> 2) * LDW + (idx & 3) * 8], &src[idx * 8]);
    }
  };

  Acc acc[2][NT];
#pragma unroll
  for (int i = 0; i < 2; i++)
#pragma unroll
    for (int j = 0; j < NT; j++)
#pragma unroll
      for (int e = 0; e < 8; e++) acc[i][j].f[e] = 0.0f;

  const int nchunks = K >> 5;

  // prologue: chunks 0 and 1 in flight; retire chunk 0 before first use
  stageA(0, 0);
  stageW(0, 0);
  if (nchunks > 1) {
    stageA(1, 32);
    stageW(1, 1);
    drain_async_n<CNT>();        // chunk 0 complete, chunk 1 may remain in flight
  } else {
    drain_async_n<0>();
  }
  __syncthreads();

  int cur = 0;
  for (int c = 0; c < nchunks; ++c) {
    // fragments per documented CDNA5 16-bit WMMA layouts
    Frag a[2];
#pragma unroll
    for (int rt = 0; rt < 2; rt++) {
      const unsigned short* p = &A_lds[cur][(r_off + rt * 16 + l15) * LDA + hl * 8];
      a[rt].q[0] = *(const int4*)(p);        // K 0..7   (+hl*8)
      a[rt].q[1] = *(const int4*)(p + 16);   // K 16..23 (+hl*8)
    }
    Frag b[NT];
#pragma unroll
    for (int ct = 0; ct < NT; ct++) {
      const unsigned short* p = &W_lds[cur][(c_off + ct * 16 + l15) * LDW + hl * 8];
      b[ct].q[0] = *(const int4*)(p);
      b[ct].q[1] = *(const int4*)(p + 16);
    }

    const bool prefetch = (c + 2 < nchunks);
    if (prefetch) {                          // stage chunk c+2, two buffers ahead
      int tgt = cur + 2; if (tgt >= 3) tgt -= 3;
      stageA(tgt, (c + 2) * 32);
      stageW(tgt, c + 2);
    }

#pragma unroll
    for (int rt = 0; rt < 2; rt++)
#pragma unroll
      for (int ct = 0; ct < NT; ct++)
        acc[rt][ct].v = __builtin_amdgcn_wmma_f32_16x16x32_bf16(
            false, a[rt].v, false, b[ct].v, (short)0, acc[rt][ct].v,
            false, false);

    // retire chunk c+1's copies (issued last iteration); keep c+2's in flight
    if (prefetch) drain_async_n<CNT>();
    else          drain_async_n<0>();
    __syncthreads();

    cur = (cur + 1 == 3) ? 0 : cur + 1;
  }

  // epilogue: bias + activation + store (C/D: VGPR i -> M=i (lanes0-15) / M=8+i)
#pragma unroll
  for (int rt = 0; rt < 2; rt++) {
#pragma unroll
    for (int ct = 0; ct < NT; ct++) {
      int n = colBlk + c_off + ct * 16 + l15;
      if (n < N) {
        float bv = bm[n];
#pragma unroll
        for (int i = 0; i < 8; i++) {
          int row = rowBlk + r_off + rt * 16 + hl * 8 + i;
          float v = acc[rt][ct].f[i] + bv;
          if (ACTK == ACT_TANH) {
            v = tanhf(v);
          } else if (ACTK == ACT_CLAMP) {
            v = 0.5f   - softplus_f(0.5f - v);    // MAX_LOG_VAR clamp
            v = -10.0f + softplus_f(v + 10.0f);   // MIN_LOG_VAR clamp
          }
          size_t oidx = ((size_t)m * BATCH + (size_t)row) * (size_t)N + (size_t)n;
          if (OUT_BF16) ((unsigned short*)outp)[oidx] = f2bf(v);
          else          ((float*)outp)[oidx] = v;
        }
      }
    }
  }
}

// ---------------------------------------------------------------------------
extern "C" void kernel_launch(void* const* d_in, const int* in_sizes, int n_in,
                              void* d_out, int out_size, void* d_ws, size_t ws_size,
                              hipStream_t stream) {
  (void)in_sizes; (void)n_in; (void)out_size; (void)ws_size;

  const float* obs   = (const float*)d_in[0];
  const float* act   = (const float*)d_in[1];
  const float* W0    = (const float*)d_in[2];
  const float* b0    = (const float*)d_in[3];
  const float* W1    = (const float*)d_in[4];
  const float* b1    = (const float*)d_in[5];
  const float* W2    = (const float*)d_in[6];
  const float* b2    = (const float*)d_in[7];
  const float* Wmu_o = (const float*)d_in[8];
  const float* bmu_o = (const float*)d_in[9];
  const float* Wmu_r = (const float*)d_in[10];
  const float* bmu_r = (const float*)d_in[11];
  const float* Wv_o  = (const float*)d_in[12];
  const float* bv_o  = (const float*)d_in[13];
  const float* Wv_r  = (const float*)d_in[14];
  const float* bv_r  = (const float*)d_in[15];

  // ---- workspace carve-up (all 256B aligned) ----
  char* ws = (char*)d_ws;
  size_t off = 0;
  auto carve = [&](size_t bytes) {
    char* p = ws + off;
    off += (bytes + 255) & ~(size_t)255;
    return p;
  };
  unsigned short* xbf = (unsigned short*)carve((size_t)BATCH * IN_D * 2);
  unsigned short* h0  = (unsigned short*)carve((size_t)M_ENS * BATCH * HID * 2);
  unsigned short* h1  = (unsigned short*)carve((size_t)M_ENS * BATCH * HID * 2);
  unsigned short* w0p = (unsigned short*)carve((size_t)M_ENS * IN_D * HID * 2);
  unsigned short* w1p = (unsigned short*)carve((size_t)M_ENS * HID * HID * 2);
  unsigned short* w2p = (unsigned short*)carve((size_t)M_ENS * HID * HID * 2);
  unsigned short* wmo = (unsigned short*)carve((size_t)M_ENS * HID * OBS_D * 2);
  unsigned short* wvo = (unsigned short*)carve((size_t)M_ENS * HID * OBS_D * 2);
  unsigned short* wmr = (unsigned short*)carve((size_t)M_ENS * HID * 32 * 2);
  unsigned short* wvr = (unsigned short*)carve((size_t)M_ENS * HID * 32 * 2);

  float* mu_o = (float*)d_out;
  float* lv_o = mu_o + (size_t)M_ENS * BATCH * OBS_D;
  float* mu_r = lv_o + (size_t)M_ENS * BATCH * OBS_D;
  float* lv_r = mu_r + (size_t)M_ENS * BATCH;

  dim3 blk(256);
  const size_t hstride = (size_t)BATCH * HID;

  // ---- phase 1: input concat->bf16 + weight packing ----
  prep_x_kernel<<<dim3((BATCH * IN_D + 255) / 256), blk, 0, stream>>>(obs, act, xbf);

  auto pack = [&](const float* W, unsigned short* Wp, int K, int N, int Npad) {
    int total = M_ENS * (K >> 5) * Npad;
    pack_w_kernel<<<dim3((total + 255) / 256), blk, 0, stream>>>(W, Wp, K, N, Npad);
  };
  pack(W0, w0p, IN_D, HID, HID);
  pack(W1, w1p, HID, HID, HID);
  pack(W2, w2p, HID, HID, HID);
  pack(Wmu_o, wmo, HID, OBS_D, OBS_D);
  pack(Wv_o,  wvo, HID, OBS_D, OBS_D);
  pack(Wmu_r, wmr, HID, 1, 32);
  pack(Wv_r,  wvr, HID, 1, 32);

  // ---- phase 2: GEMMs ----
  dim3 gHid(BATCH / 128, HID / 128, M_ENS);
  gemm_wmma_kernel<4, ACT_TANH, true><<<gHid, blk, 0, stream>>>(
      xbf, 0, w0p, b0, h0, IN_D, HID, HID);                // x shared across m
  gemm_wmma_kernel<4, ACT_TANH, true><<<gHid, blk, 0, stream>>>(
      h0, hstride, w1p, b1, h1, HID, HID, HID);
  gemm_wmma_kernel<4, ACT_TANH, true><<<gHid, blk, 0, stream>>>(
      h1, hstride, w2p, b2, h0, HID, HID, HID);

  dim3 gHead(BATCH / 128, 1, M_ENS);
  gemm_wmma_kernel<2, ACT_ID,    false><<<gHead, blk, 0, stream>>>(
      h0, hstride, wmo, bmu_o, mu_o, HID, OBS_D, OBS_D);
  gemm_wmma_kernel<2, ACT_CLAMP, false><<<gHead, blk, 0, stream>>>(
      h0, hstride, wvo, bv_o,  lv_o, HID, OBS_D, OBS_D);
  gemm_wmma_kernel<1, ACT_ID,    false><<<gHead, blk, 0, stream>>>(
      h0, hstride, wmr, bmu_r, mu_r, HID, 1, 32);
  gemm_wmma_kernel<1, ACT_CLAMP, false><<<gHead, blk, 0, stream>>>(
      h0, hstride, wvr, bv_r,  lv_r, HID, 1, 32);
}
